// GeometricAttention_13280038879497
// MI455X (gfx1250) — matrix operand order repaired and checked by reading, not compiled, not run on hardware
//
#include <hip/hip_runtime.h>
#include <hip/hip_bf16.h>
#include <math.h>

#define B_   2
#define N_   1024
#define DIM_ 64
#define HID_ 128
#define BPAD (HID_ + 8)   // f16 elements per padded LDS row (272B: kills 64-bank conflicts, keeps 16B align)

typedef _Float16 v16h __attribute__((ext_vector_type(16)));
typedef float    v8f  __attribute__((ext_vector_type(8)));
typedef _Float16 h2   __attribute__((ext_vector_type(2)));
typedef int      v4i  __attribute__((vector_size(16)));   // matches async-builtin param type

#if __has_builtin(__builtin_amdgcn_global_load_async_to_lds_b128)
#define HAS_ASYNC 1
#else
#define HAS_ASYNC 0
#endif

static __device__ inline h2 h2splat(float f) {
  h2 r; r.x = (_Float16)f; r.y = (_Float16)f; return r;
}

static __device__ inline void wait_async1() {
#if HAS_ASYNC
#if __has_builtin(__builtin_amdgcn_s_wait_asynccnt)
  __builtin_amdgcn_s_wait_asynccnt(1);
#else
  asm volatile("s_wait_asynccnt 1" ::: "memory");
#endif
#endif
}

static __device__ inline void wait_async0() {
#if HAS_ASYNC
#if __has_builtin(__builtin_amdgcn_s_wait_asynccnt)
  __builtin_amdgcn_s_wait_asynccnt(0);
#else
  asm volatile("s_wait_asynccnt 0" ::: "memory");
#endif
#endif
}

// ---- CDNA5 wave32 WMMA fragment loaders (16x16x32 f16) ----
// A-matrix 16x32 (MxK): lane L (half = L>>4, m = L&15):
//   element e holds K = e + 16*half (matches B fragment)
static __device__ inline v16h load_a_frag(const float* src, int m0, int ld, int k0) {
  const int lane = threadIdx.x & 31;
  const int hl = lane >> 4;
  const int m = m0 + (lane & 15);
  v16h a;
#pragma unroll
  for (int e = 0; e < 16; ++e) {
    const int v = e >> 1, p = e & 1;
    const int k = 2 * v + p + 16 * hl;
    a[e] = (_Float16)src[(size_t)m * ld + (k0 + k)];
  }
  return a;
}

// B-matrix 32x16 (KxN): lanes 0-15 hold K=0..15 (col = lane&15), lanes 16-31 hold K=16..31.
static __device__ inline v16h load_b_frag(const float* src, int k0, int ldb, int n0) {
  const int lane = threadIdx.x & 31;
  const int n = n0 + (lane & 15);
  const int kh = (lane >> 4) * 16;
  v16h b;
#pragma unroll
  for (int e = 0; e < 16; ++e) {
    const int v = e >> 1, p = e & 1;
    const int k = 2 * v + p + kh;
    b[e] = (_Float16)src[(size_t)(k0 + k) * ldb + n];
  }
  return b;
}

// ---------------------------------------------------------------------------
// Kernel 1: A = h@W1a + b1 and Bv = h@W1b, stored as f16 (2048x128 each).
// One wave per 16x16 output tile; K=64 -> 2 WMMA per tile.
// ---------------------------------------------------------------------------
__global__ void __launch_bounds__(256)
precompute_AB_wmma(const float* __restrict__ h,
                   const float* __restrict__ W1,
                   const float* __restrict__ b1,
                   _Float16* __restrict__ Af16,
                   _Float16* __restrict__ Bf16) {
  const int wave = threadIdx.x >> 5;
  const int task = blockIdx.x * 8 + wave;   // 0..2047
  const int which = task >> 10;             // 0: A (W1a, +b1), 1: Bv (W1b)
  const int t = task & 1023;
  const int mt = t >> 3, nt = t & 7;        // 128 row tiles x 8 col tiles
  const int m0 = mt * 16, n0 = nt * 16;
  const int lane = threadIdx.x & 31;

  v8f c = {};
#pragma unroll
  for (int kc = 0; kc < DIM_; kc += 32) {
    v16h a = load_a_frag(h, m0, DIM_, kc);
    v16h b = load_b_frag(W1 + (size_t)which * DIM_ * HID_, kc, HID_, n0);
    c = __builtin_amdgcn_wmma_f32_16x16x32_f16(false, a, false, b, (short)0, c,
                                               false, false);
  }

  _Float16* dst = which ? Bf16 : Af16;
  const int n = n0 + (lane & 15);
  const float add = which ? 0.0f : b1[n];
#pragma unroll
  for (int r = 0; r < 8; ++r) {
    const int m = m0 + r + 8 * (lane >> 4);
    dst[(size_t)m * HID_ + n] = (_Float16)(c[r] + add);
  }
}

// ---------------------------------------------------------------------------
// Kernel 2: the N^2 hot loop. One block per (batch, 16-row i tile).
// Bv j-tiles (16 rows x 256B) are double-buffered into LDS with
// GLOBAL_LOAD_ASYNC_TO_LDS_B128 (ASYNCcnt), overlapping DMA with the
// packed-f16 hid + polynomial-erf GELU + packed dot with W2.
// ---------------------------------------------------------------------------
__global__ void __launch_bounds__(256)
pairwise_kernel(const float* __restrict__ x,
                const unsigned char* __restrict__ mask,
                const _Float16* __restrict__ Af16,
                const _Float16* __restrict__ Bf16,
                const float* __restrict__ W1,
                const float* __restrict__ W2,
                const float* __restrict__ b2,
                float* __restrict__ dx_out,
                float* __restrict__ hagg) {
  __shared__ _Float16 sA[16 * HID_];
  __shared__ _Float16 sB[2 * 16 * BPAD];     // double-buffered Bv j-tile
  __shared__ _Float16 sW1c[HID_];
  __shared__ _Float16 sW2[HID_];
  __shared__ float sx[16][3];
  __shared__ float sacc[16][4];

  const int tid = threadIdx.x;
  const int b = blockIdx.x >> 6;
  const int i0 = (blockIdx.x & 63) * 16;
  const int rowbase = b * N_ + i0;

  for (int idx = tid; idx < 16 * HID_; idx += 256)
    sA[idx] = Af16[(size_t)(rowbase + (idx >> 7)) * HID_ + (idx & (HID_ - 1))];
  if (tid < HID_) {
    sW1c[tid] = (_Float16)W1[(size_t)2 * DIM_ * HID_ + tid];  // W1 row 128
    sW2[tid]  = (_Float16)W2[tid];
  }
  if (tid < 48) sx[tid / 3][tid % 3] = x[(size_t)(rowbase + tid / 3) * 3 + tid % 3];
  if (tid < 64) ((float*)sacc)[tid] = 0.0f;

  // ---- staging: each thread moves one 16B segment of the 4KB j-tile ----
  const int srow = tid >> 4;   // 0..15  (j row within tile)
  const int sseg = tid & 15;   // 0..15  (16B segment within 256B row)
  const _Float16* gbase = Bf16 + (size_t)b * N_ * HID_ + sseg * 8;
  _Float16* lbase = sB + srow * BPAD + sseg * 8;

#define STAGE_TILE(jt_)                                                        \
  do {                                                                         \
    const _Float16* gsrc = gbase + (size_t)((jt_)*16 + srow) * HID_;           \
    _Float16* ldst = lbase + (((jt_)&1) ? 16 * BPAD : 0);                      \
    if (HAS_ASYNC) {                                                           \
      __builtin_amdgcn_global_load_async_to_lds_b128(                          \
          (__attribute__((address_space(1))) v4i*)gsrc,                        \
          (__attribute__((address_space(3))) v4i*)ldst, 0, 0);                 \
    } else {                                                                   \
      *(float4*)ldst = *(const float4*)gsrc;                                   \
    }                                                                          \
  } while (0)

  STAGE_TILE(0);

  const int il = tid >> 4;
  const int jl = tid & 15;
  const float ax = sx[il][0], ay = sx[il][1], az = sx[il][2];
  const h2* a2   = (const h2*)(sA + il * HID_);
  const h2* w1c2 = (const h2*)sW1c;
  const h2* w22  = (const h2*)sW2;
  const float b2v = b2[0];

  float accx = 0.0f, accy = 0.0f, accz = 0.0f, acch = 0.0f;

  for (int jt = 0; jt < N_ / 16; ++jt) {
    if (jt + 1 < N_ / 16) {
      STAGE_TILE(jt + 1);
      wait_async1();
    } else {
      wait_async0();
    }
    __syncthreads();   // tile jt resident in LDS for everyone

    const int j = jt * 16 + jl;
    const size_t jrow = (size_t)(b * N_ + j);
    const float jx = x[jrow * 3 + 0];
    const float jy = x[jrow * 3 + 1];
    const float jz = x[jrow * 3 + 2];
    const float dxv = ax - jx, dyv = ay - jy, dzv = az - jz;
    const float sq = dxv * dxv + dyv * dyv + dzv * dzv;
    const float dist = sq > 0.0f ? sqrtf(sq) : 0.0f;
    const float inv = 1.0f / (dist + 1e-8f);

    const h2* bv = (const h2*)(sB + ((jt & 1) ? 16 * BPAD : 0) + jl * BPAD);

    const h2 d2 = h2splat(dist);
    h2 acc = h2splat(0.0f);
#pragma unroll 8
    for (int hh = 0; hh < HID_ / 2; ++hh) {
      h2 tv = a2[hh] + bv[hh] + d2 * w1c2[hh];
      // gelu(t) = 0.5 t (1 + erf(t/sqrt2)); erf via odd poly, clamped to [-1,1]
      h2 z  = tv * h2splat(0.70710678f);
      h2 z2 = z * z;
      h2 p  = h2splat(-0.0268661706f);
      p = p * z2 + h2splat(0.1128379167f);
      p = p * z2 + h2splat(-0.3761263890f);
      p = p * z2 + h2splat(1.1283791671f);
      h2 erfz = z * p;
      erfz = __builtin_elementwise_min(
                 __builtin_elementwise_max(erfz, h2splat(-1.0f)), h2splat(1.0f));
      h2 g = tv * h2splat(0.5f) * (h2splat(1.0f) + erfz);
      acc += g * w22[hh];
    }
    float w = (float)acc.x + (float)acc.y + b2v;
    if (!mask[jrow]) w = 0.0f;
    accx += w * dxv * inv;
    accy += w * dyv * inv;
    accz += w * dzv * inv;
    acch += w * dist;

    __syncthreads();   // everyone done with buffer (jt&1) before it is re-staged
  }

  atomicAdd(&sacc[il][0], accx);
  atomicAdd(&sacc[il][1], accy);
  atomicAdd(&sacc[il][2], accz);
  atomicAdd(&sacc[il][3], acch);
  __syncthreads();

  if (tid < 16) {
    const int row = rowbase + tid;
    dx_out[(size_t)row * 3 + 0] = sacc[tid][0];
    dx_out[(size_t)row * 3 + 1] = sacc[tid][1];
    dx_out[(size_t)row * 3 + 2] = sacc[tid][2];
    hagg[row] = sacc[tid][3];
  }
#undef STAGE_TILE
}

// ---------------------------------------------------------------------------
// Kernel 3: z = [h | h_agg] @ W3 + b3 ; LayerNorm -> h_new.
// 4 waves/block, one 16-row tile per wave, WMMA over K=64 + rank-1 h_agg term.
// ---------------------------------------------------------------------------
__global__ void __launch_bounds__(128)
output_kernel(const float* __restrict__ h,
              const float* __restrict__ W3,
              const float* __restrict__ b3,
              const float* __restrict__ ln_g,
              const float* __restrict__ ln_b,
              const float* __restrict__ hagg,
              float* __restrict__ hnew_out) {
  __shared__ float zbuf[64][DIM_ + 1];
  const int tid = threadIdx.x;
  const int lane = tid & 31;
  const int wave = tid >> 5;
  const int rowbase = blockIdx.x * 64;
  const int m0 = rowbase + wave * 16;

#pragma unroll
  for (int nt = 0; nt < 4; ++nt) {
    const int n0 = nt * 16;
    v8f c = {};
#pragma unroll
    for (int kc = 0; kc < DIM_; kc += 32) {
      v16h a   = load_a_frag(h, m0, DIM_, kc);
      v16h bfr = load_b_frag(W3, kc, DIM_, n0);
      c = __builtin_amdgcn_wmma_f32_16x16x32_f16(false, a, false, bfr, (short)0,
                                                 c, false, false);
    }
    const int n = n0 + (lane & 15);
#pragma unroll
    for (int r = 0; r < 8; ++r) {
      const int m = m0 + r + 8 * (lane >> 4);
      zbuf[m - rowbase][n] = c[r] + hagg[m] * W3[(size_t)DIM_ * DIM_ + n] + b3[n];
    }
  }
  __syncthreads();

  if (tid < 64) {
    const int row = rowbase + tid;
    float mu = 0.0f;
#pragma unroll
    for (int n = 0; n < DIM_; ++n) mu += zbuf[tid][n];
    mu *= (1.0f / DIM_);
    float var = 0.0f;
#pragma unroll
    for (int n = 0; n < DIM_; ++n) { const float d = zbuf[tid][n] - mu; var += d * d; }
    var *= (1.0f / DIM_);
    const float rs = rsqrtf(var + 1e-5f);
#pragma unroll
    for (int n = 0; n < DIM_; ++n)
      hnew_out[(size_t)row * DIM_ + n] = (zbuf[tid][n] - mu) * rs * ln_g[n] + ln_b[n];
  }
}

// ---------------------------------------------------------------------------
extern "C" void kernel_launch(void* const* d_in, const int* in_sizes, int n_in,
                              void* d_out, int out_size, void* d_ws, size_t ws_size,
                              hipStream_t stream) {
  const float* x    = (const float*)d_in[0];
  const float* h    = (const float*)d_in[1];
  const unsigned char* mask = (const unsigned char*)d_in[2];
  const float* W1   = (const float*)d_in[3];
  const float* b1   = (const float*)d_in[4];
  const float* W2   = (const float*)d_in[5];
  const float* b2   = (const float*)d_in[6];
  const float* W3   = (const float*)d_in[7];
  const float* b3   = (const float*)d_in[8];
  const float* ln_g = (const float*)d_in[9];
  const float* ln_b = (const float*)d_in[10];

  char* ws = (char*)d_ws;
  _Float16* Af16 = (_Float16*)ws;                 // 2048*128*2 = 512 KB
  _Float16* Bf16 = (_Float16*)(ws + 512 * 1024);  // 512 KB
  float*    hagg = (float*)(ws + 1024 * 1024);    // 8 KB

  float* dx_out   = (float*)d_out;                // B*N*3 floats
  float* hnew_out = dx_out + (size_t)B_ * N_ * 3; // B*N*DIM floats

  // 2048 wave-tasks (A and Bv tiles), 8 waves/block.
  precompute_AB_wmma<<<256, 256, 0, stream>>>(h, W1, b1, Af16, Bf16);
  // One block per (batch, 16-i tile): 128 blocks x 256 threads.
  pairwise_kernel<<<B_ * (N_ / 16), 256, 0, stream>>>(x, mask, Af16, Bf16, W1, W2,
                                                      b2, dx_out, hagg);
  // 32 blocks x 128 threads cover all 2048 rows.
  output_kernel<<<(B_ * N_) / 64, 128, 0, stream>>>(h, W3, b3, ln_g, ln_b, hagg,
                                                    hnew_out);
}
